// MushroomBodyLayer_32865089749508
// MI455X (gfx1250) — matrix operand order repaired and checked by reading, not compiled, not run on hardware
//
#include <hip/hip_runtime.h>

#define BATCH   4096
#define IN_DIM  256
#define UNITS   16384
#define TOPK    1638

#define MT 64      // M tile per block
#define NT 128     // N tile per block
#define LDA 264    // f16 elems per A row in LDS (256 + 8 pad)
#define LDB 264    // f16 elems per Bt row in LDS

typedef __attribute__((ext_vector_type(4)))  _Float16 v4h;
typedef __attribute__((ext_vector_type(8)))  _Float16 v8h;
typedef __attribute__((ext_vector_type(16))) _Float16 v16h;
typedef __attribute__((ext_vector_type(8)))  float    v8f;

// ---------------------------------------------------------------------------
// Preprocess 1: x (f32, [BATCH][IN_DIM]) -> Xh (f16, same layout). Done once
// per call instead of once per tile-restage.
// ---------------------------------------------------------------------------
__global__ __launch_bounds__(256) void mb_convert_x(
    const float* __restrict__ x, _Float16* __restrict__ Xh)
{
  int idx = blockIdx.x * 256 + threadIdx.x;           // one float4 each
  const int total = BATCH * IN_DIM / 4;
  if (idx < total) {
    float4 v = ((const float4*)x)[idx];
    v4h h = { (_Float16)v.x, (_Float16)v.y, (_Float16)v.z, (_Float16)v.w };
    ((v4h*)Xh)[idx] = h;
  }
}

// ---------------------------------------------------------------------------
// Preprocess 2: W (f32, [IN_DIM][UNITS]) -> Wt (f16, [UNITS][IN_DIM]),
// LDS-tiled 64x64 transpose so both global sides are coalesced.
// ---------------------------------------------------------------------------
__global__ __launch_bounds__(256) void mb_transpose_w(
    const float* __restrict__ W, _Float16* __restrict__ Wt)
{
  __shared__ _Float16 tile[64][72];                   // +8 pad vs bank conflicts
  const int k0 = blockIdx.x * 64;                     // IN_DIM/64 = 4
  const int n0 = blockIdx.y * 64;                     // UNITS/64 = 256
  const int tid = threadIdx.x;

  for (int i = tid; i < 64 * 16; i += 256) {          // 4096 elems, float4 reads
    int r = i >> 4, c4 = i & 15;
    float4 v = ((const float4*)(W + (size_t)(k0 + r) * UNITS + n0))[c4];
    int c = c4 << 2;
    tile[r][c + 0] = (_Float16)v.x;
    tile[r][c + 1] = (_Float16)v.y;
    tile[r][c + 2] = (_Float16)v.z;
    tile[r][c + 3] = (_Float16)v.w;
  }
  __syncthreads();
  for (int i = tid; i < 64 * 8; i += 256) {           // 16B coalesced writes
    int j = i >> 3, c8 = (i & 7) << 3;
    v8h h;
    #pragma unroll
    for (int s = 0; s < 8; ++s) h[s] = tile[c8 + s][j];
    *(v8h*)&Wt[(size_t)(n0 + j) * IN_DIM + k0 + c8] = h;
  }
}

// ---------------------------------------------------------------------------
// Kernel 3: h = relu(Xh @ Wt^T + bias) via f16 WMMA. Tiles staged into LDS
// with CDNA5 async global->LDS copies (ASYNCcnt), no VGPR round-trip.
// Block = 256 threads (8 wave32). Tile = 64(M) x 128(N), full K=256 in LDS.
// ---------------------------------------------------------------------------
__global__ __launch_bounds__(256) void mb_gemm_relu(
    const _Float16* __restrict__ Xh, const _Float16* __restrict__ Wt,
    const float* __restrict__ bias, float* __restrict__ out)
{
  extern __shared__ char smem[];
  _Float16* As = (_Float16*)smem;        // [MT][LDA]  row-major (M x K)
  _Float16* Bt = As + MT * LDA;          // [NT][LDB]  (N x K), contiguous K

  const int tid = threadIdx.x;
  const int m0  = blockIdx.y * MT;
  const int n0  = blockIdx.x * NT;

  // LDS byte offsets (low 32 bits of the generic shared-aperture address)
  const unsigned asBase = (unsigned)(uintptr_t)As;
  const unsigned btBase = (unsigned)(uintptr_t)Bt;

  // ---- async stage A tile: 64 rows x 256 f16, 16B chunks per lane
  for (int i = tid; i < MT * (IN_DIM / 8); i += 256) {   // 2048 chunks
    int r = i >> 5, c = (i & 31) << 3;
    unsigned lds = asBase + (unsigned)((r * LDA + c) * 2);
    const _Float16* g = Xh + (size_t)(m0 + r) * IN_DIM + c;
    asm volatile("global_load_async_to_lds_b128 %0, %1, off"
                 :: "v"(lds), "v"(g) : "memory");
  }
  // ---- async stage B tile: 128 rows x 256 f16 from pre-transposed Wt
  for (int i = tid; i < NT * (IN_DIM / 8); i += 256) {   // 4096 chunks
    int r = i >> 5, c = (i & 31) << 3;
    unsigned lds = btBase + (unsigned)((r * LDB + c) * 2);
    const _Float16* g = Wt + (size_t)(n0 + r) * IN_DIM + c;
    asm volatile("global_load_async_to_lds_b128 %0, %1, off"
                 :: "v"(lds), "v"(g) : "memory");
  }
  asm volatile("s_wait_asynccnt 0" ::: "memory");
  __syncthreads();

  const int lane = tid & 31;
  const int wave = tid >> 5;
  const int m16  = (wave & 3) * 16;   // 4 M-tiles across waves
  const int ng   = (wave >> 2) * 64;  // 2 N-groups of 64 across waves
  const int ln   = lane & 15;
  const int half = lane >> 4;         // 0 or 1 (which half-wave)

  v8f acc[4] = {{}, {}, {}, {}};

  // A fragment per lane: row M = m16+ln; K chunks at k0+{0,16} + half*8
  const _Float16* arow = As + (m16 + ln) * LDA + half * 8;

  #pragma unroll
  for (int k0 = 0; k0 < IN_DIM; k0 += 32) {
    v8h a0 = *(const v8h*)(arow + k0);
    v8h a1 = *(const v8h*)(arow + k0 + 16);
    v16h a = __builtin_shufflevector(a0, a1,
        0,1,2,3,4,5,6,7,8,9,10,11,12,13,14,15);
    #pragma unroll
    for (int t = 0; t < 4; ++t) {
      // B fragment per lane: column N = ng+t*16+ln; K chunks at
      // k0+{0,16} + half*8 (same order as A)
      const _Float16* bp = Bt + (ng + t * 16 + ln) * LDB + k0 + half * 8;
      v8h b0 = *(const v8h*)bp;
      v8h b1 = *(const v8h*)(bp + 16);
      v16h b = __builtin_shufflevector(b0, b1,
          0,1,2,3,4,5,6,7,8,9,10,11,12,13,14,15);
      acc[t] = __builtin_amdgcn_wmma_f32_16x16x32_f16(
          false, a, false, b, (short)0, acc[t], false, false);
    }
  }

  // Epilogue: C layout -> VGPR r: lanes 0-15 M=r, lanes 16-31 M=8+r; N=ln
  #pragma unroll
  for (int t = 0; t < 4; ++t) {
    int gn = n0 + ng + t * 16 + ln;
    float bv = bias[gn];
    int gm = m0 + m16 + half * 8;
    float* op = out + (size_t)gm * UNITS + gn;
    #pragma unroll
    for (int r = 0; r < 8; ++r) {
      float v = acc[t][r] + bv;
      op[(size_t)r * UNITS] = v > 0.0f ? v : 0.0f;
    }
  }
}

// ---------------------------------------------------------------------------
// Kernel 4: per-row exact top-K via 4-pass radix select on order-preserving
// uint keys; in-place threshold with tie budget. One block per row.
// ---------------------------------------------------------------------------
__device__ __forceinline__ unsigned f2k(float f) {
  unsigned u = __float_as_uint(f);
  return (u & 0x80000000u) ? ~u : (u | 0x80000000u);
}

__global__ __launch_bounds__(256) void mb_topk(float* __restrict__ out)
{
  extern __shared__ char smem[];
  float*    vals = (float*)smem;                 // [UNITS]
  unsigned* hist = (unsigned*)(vals + UNITS);    // [256]
  unsigned* ctrl = hist + 256;                   // [0]=prefix [1]=remaining [2]=tie

  const int row = blockIdx.x;
  const int tid = threadIdx.x;
  float* rp = out + (size_t)row * UNITS;

  for (int i = tid; i < UNITS; i += 256) vals[i] = rp[i];
  if (tid == 0) { ctrl[0] = 0u; ctrl[1] = TOPK; }
  __syncthreads();

  #pragma unroll 1
  for (int pass = 0; pass < 4; ++pass) {
    const int shift = 24 - 8 * pass;
    for (int d = tid; d < 256; d += 256) hist[d] = 0u;
    __syncthreads();
    const unsigned prefix = ctrl[0];
    const unsigned mask = pass ? (0xFFFFFFFFu << (shift + 8)) : 0u;
    for (int i = tid; i < UNITS; i += 256) {
      unsigned key = f2k(vals[i]);
      if ((key & mask) == prefix)
        atomicAdd(&hist[(key >> shift) & 255u], 1u);
    }
    __syncthreads();
    if (tid == 0) {
      unsigned rem = ctrl[1], cum = 0u;
      int d;
      for (d = 255; d > 0; --d) {
        unsigned h = hist[d];
        if (cum + h >= rem) break;
        cum += h;
      }
      ctrl[1] = rem - cum;
      ctrl[0] = prefix | ((unsigned)d << shift);
    }
    __syncthreads();
  }

  const unsigned T = ctrl[0];         // key of the K-th largest value
  const unsigned tiesKeep = ctrl[1];  // how many elements equal to T to keep
  if (tid == 0) ctrl[2] = 0u;
  __syncthreads();

  for (int i = tid; i < UNITS; i += 256) {
    float v = vals[i];
    unsigned key = f2k(v);
    float o = 0.0f;
    if (key > T) {
      o = v;
    } else if (key == T) {
      unsigned p = atomicAdd(&ctrl[2], 1u);
      if (p < tiesKeep) o = v;
    }
    rp[i] = o;
  }
}

// ---------------------------------------------------------------------------
extern "C" void kernel_launch(void* const* d_in, const int* in_sizes, int n_in,
                              void* d_out, int out_size, void* d_ws, size_t ws_size,
                              hipStream_t stream) {
  (void)in_sizes; (void)n_in; (void)out_size; (void)ws_size;
  const float* x    = (const float*)d_in[0];
  const float* W    = (const float*)d_in[1];
  const float* bias = (const float*)d_in[2];
  float* out = (float*)d_out;

  // Scratch layout: Xh (2 MB f16) then Wt (8 MB f16, transposed)
  _Float16* Xh = (_Float16*)d_ws;
  _Float16* Wt = Xh + (size_t)BATCH * IN_DIM;

  const size_t sh1 = (size_t)(MT * LDA + NT * LDB) * sizeof(_Float16); // ~99 KB
  const size_t sh2 = (size_t)UNITS * sizeof(float) + 260 * sizeof(unsigned);

  // CDNA5 WGP has 320 KB LDS; raise the opt-in limit (no-op where unneeded).
  (void)hipFuncSetAttribute((const void*)mb_gemm_relu,
      hipFuncAttributeMaxDynamicSharedMemorySize, (int)sh1);
  (void)hipFuncSetAttribute((const void*)mb_topk,
      hipFuncAttributeMaxDynamicSharedMemorySize, (int)sh2);

  mb_convert_x<<<BATCH * IN_DIM / 4 / 256, 256, 0, stream>>>(x, Xh);
  dim3 gt(IN_DIM / 64, UNITS / 64);                  // 4 x 256
  mb_transpose_w<<<gt, 256, 0, stream>>>(W, Wt);
  dim3 g1(UNITS / NT, BATCH / MT);                   // 128 x 64 blocks
  mb_gemm_relu<<<g1, 256, sh1, stream>>>(Xh, Wt, bias, out);
  mb_topk<<<BATCH, 256, sh2, stream>>>(out);
}
